// TopKSAE_84610855731333
// MI455X (gfx1250) — compile-verified, matrix operand order checked by reading, not run on hardware
//
#include <hip/hip_runtime.h>
#include <hip/hip_bf16.h>

typedef __attribute__((ext_vector_type(16))) __bf16 v16bf;
typedef __attribute__((ext_vector_type(8)))  float  v8f;

#define D_IN     768
#define D_LATENT 16384
#define BATCH    8192
#define TOPK     32

#define BM 128
#define BN 128
#define BK 32
#define KSTEPS (D_IN / BK)   // 24
#define LDSS 40   // bf16 elements per LDS tile row: 32 + 8 pad (16B) -> conflict-free

// 32-bit LDS byte offset of a shared-memory pointer (for async-to-LDS VDST operand)
__device__ __forceinline__ unsigned lds_off(const void* p) {
    return (unsigned)(unsigned long long)(__attribute__((address_space(3))) const void*)p;
}

// ---------------- fp32 -> bf16 conversion (grid-stride) ----------------
__global__ void cvt_f32_bf16(const float* __restrict__ src, __bf16* __restrict__ dst, int n) {
    int i = blockIdx.x * blockDim.x + threadIdx.x;
    int stride = gridDim.x * blockDim.x;
    for (; i < n; i += stride) dst[i] = (__bf16)src[i];
}

// ---------------- encoder: z = relu(x @ W_enc^T), bf16 WMMA + async double buffer ----------------
__global__ __launch_bounds__(256)
void sae_encoder_wmma(const __bf16* __restrict__ X,   // [BATCH, D_IN] bf16
                      const __bf16* __restrict__ W,   // [D_LATENT, D_IN] bf16
                      float* __restrict__ Z)          // [BATCH, D_LATENT] f32
{
    __shared__ __bf16 As[2][BM * LDSS];
    __shared__ __bf16 Bs[2][BN * LDSS];

    const int t    = threadIdx.x;
    const int lane = t & 31;
    const int wave = t >> 5;        // 0..7
    const int wm   = wave & 3;      // M sub-tile (32 rows each)
    const int wn   = wave >> 2;     // N sub-tile (64 cols each)
    const int half = lane >> 4;     // 0: lanes 0-15, 1: lanes 16-31
    const int l15  = lane & 15;

    const int mBlock = blockIdx.y * BM;
    const int nBlock = blockIdx.x * BN;

    v8f acc[2][4];
#pragma unroll
    for (int i = 0; i < 2; ++i)
#pragma unroll
        for (int j = 0; j < 4; ++j)
#pragma unroll
            for (int r = 0; r < 8; ++r) acc[i][j][r] = 0.0f;

    // tile staging coords: each thread copies 32 bf16 (two 16B chunks for A, two for B)
    const int ldRow = t >> 1;            // 0..127
    const int ldCol = (t & 1) * 16;      // 0 or 16

    const __bf16* xp = X + (size_t)(mBlock + ldRow) * D_IN + ldCol;
    const __bf16* wp = W + (size_t)(nBlock + ldRow) * D_IN + ldCol;

    const unsigned slot = (unsigned)(ldRow * LDSS + ldCol) * 2u;  // byte offset in tile
    unsigned ldsA[2] = { lds_off(&As[0][0]) + slot, lds_off(&As[1][0]) + slot };
    unsigned ldsB[2] = { lds_off(&Bs[0][0]) + slot, lds_off(&Bs[1][0]) + slot };

    // issue one stage of async memory->LDS copies (16B x2 for A, 16B x2 for B)
    auto stage_load = [&](int s, int buf) {
        unsigned long long ga = (unsigned long long)(xp + s * BK);
        unsigned long long gb = (unsigned long long)(wp + s * BK);
        asm volatile(
            "global_load_async_to_lds_b128 %0, %2, off\n\t"
            "global_load_async_to_lds_b128 %0, %2, off offset:16\n\t"
            "global_load_async_to_lds_b128 %1, %3, off\n\t"
            "global_load_async_to_lds_b128 %1, %3, off offset:16"
            :
            : "v"(ldsA[buf]), "v"(ldsB[buf]), "v"(ga), "v"(gb)
            : "memory");
    };

    // per the ISA 16-bit A/B layout: lanes 0-15 hold K 0-7 & 16-23, lanes 16-31 hold K 8-15 & 24-31
    const int c0 = half ? 8  : 0;
    const int c1 = half ? 24 : 16;

    // prologue: fill buffer 0
    stage_load(0, 0);
    asm volatile("s_wait_asynccnt 0" ::: "memory");
    __syncthreads();

    for (int s = 0; s < KSTEPS; ++s) {
        const int cur = s & 1;

        // overlap: start next stage's copy into the other buffer
        if (s + 1 < KSTEPS) stage_load(s + 1, cur ^ 1);

        const __bf16* Ab = As[cur];
        const __bf16* Bb = Bs[cur];

        v16bf afrag[2], bfrag[4];
#pragma unroll
        for (int i = 0; i < 2; ++i) {
            int r = wm * 32 + i * 16 + l15;
            union { v16bf v; uint4 q[2]; } u;
            u.q[0] = *(const uint4*)&Ab[r * LDSS + c0];
            u.q[1] = *(const uint4*)&Ab[r * LDSS + c1];
            afrag[i] = u.v;
        }
#pragma unroll
        for (int j = 0; j < 4; ++j) {
            int r = wn * 64 + j * 16 + l15;
            union { v16bf v; uint4 q[2]; } u;
            u.q[0] = *(const uint4*)&Bb[r * LDSS + c0];
            u.q[1] = *(const uint4*)&Bb[r * LDSS + c1];
            bfrag[j] = u.v;
        }
#pragma unroll
        for (int i = 0; i < 2; ++i)
#pragma unroll
            for (int j = 0; j < 4; ++j)
                acc[i][j] = __builtin_amdgcn_wmma_f32_16x16x32_bf16(
                    false, afrag[i], false, bfrag[j], (short)0, acc[i][j], false, false);

        if (s + 1 < KSTEPS) {
            asm volatile("s_wait_asynccnt 0" ::: "memory");
            __syncthreads();
        }
    }

    // epilogue: ReLU + store (C/D layout: VGPR r -> M = r + 8*half, N = lane&15)
#pragma unroll
    for (int i = 0; i < 2; ++i) {
#pragma unroll
        for (int j = 0; j < 4; ++j) {
            int mbase = mBlock + wm * 32 + i * 16 + half * 8;
            int n     = nBlock + wn * 64 + j * 16 + l15;
            float* zp = Z + (size_t)mbase * D_LATENT + n;
#pragma unroll
            for (int r = 0; r < 8; ++r) {
                float v = acc[i][j][r];
                zp[(size_t)r * D_LATENT] = v > 0.0f ? v : 0.0f;
            }
        }
    }
}

// ---------------- per-row radix top-k (K=32 of 16384), mask z in place ----------------
__global__ __launch_bounds__(256)
void sae_topk(float* __restrict__ Z, int* __restrict__ topIdx, float* __restrict__ topVal) {
    const int row = blockIdx.x;
    float* zrow = Z + (size_t)row * D_LATENT;
    const int t = threadIdx.x;

    __shared__ int hist[256];
    __shared__ unsigned sPrefix;
    __shared__ int sKRemain;
    __shared__ int selCount, tieCount;

    if (t == 0) { sPrefix = 0u; sKRemain = TOPK; selCount = 0; tieCount = 0; }
    if (t < TOPK) { topIdx[row * TOPK + t] = 0; topVal[row * TOPK + t] = 0.0f; }
    __syncthreads();

    // z >= 0 after ReLU, so raw float bits are order-preserving unsigned keys.
    for (int pass = 0; pass < 4; ++pass) {
        const int shift = 24 - 8 * pass;
        hist[t] = 0;
        __syncthreads();
        const unsigned prefix = sPrefix;
        const unsigned hiMask = (pass == 0) ? 0u : (0xFFFFFFFFu << (shift + 8));
        for (int i = t; i < D_LATENT; i += 256) {
            unsigned key = __float_as_uint(zrow[i]);
            if ((key & hiMask) == (prefix & hiMask))
                atomicAdd(&hist[(key >> shift) & 255], 1);
        }
        __syncthreads();
        if (t == 0) {
            int need = sKRemain;
            int cum = 0, b = 255;
            for (; b > 0; --b) {
                cum += hist[b];
                if (cum >= need) break;
            }
            if (cum < need) cum += hist[0];           // fell through to bin 0
            sKRemain = need - (cum - hist[b]);        // still needed among ties in bin b
            sPrefix  = sPrefix | ((unsigned)b << shift);
        }
        __syncthreads();
    }

    const unsigned thr = sPrefix;     // key of the K-th largest element
    const int tieQuota = sKRemain;    // how many threshold-equal elements to keep
    __syncthreads();

    for (int i = t; i < D_LATENT; i += 256) {
        float v = zrow[i];
        unsigned key = __float_as_uint(v);
        bool keep = false;
        if (key > thr) {
            keep = true;
        } else if (key == thr) {
            int tk = atomicAdd(&tieCount, 1);
            keep = (tk < tieQuota);
        }
        if (keep) {
            int slot = atomicAdd(&selCount, 1);
            if (slot < TOPK) {
                topIdx[row * TOPK + slot] = i;
                topVal[row * TOPK + slot] = v;
            }
        } else {
            zrow[i] = 0.0f;
        }
    }
}

// ---------------- sparse decoder: x_hat[b,:] = sum_j val_j * W_dec[:, idx_j] ----------------
__global__ __launch_bounds__(256)
void sae_decoder(const float* __restrict__ Wd,      // [D_IN, D_LATENT] f32
                 const int* __restrict__ topIdx,
                 const float* __restrict__ topVal,
                 float* __restrict__ Xhat)          // [BATCH, D_IN]
{
    const int row = blockIdx.x;
    const int t = threadIdx.x;
    __shared__ int   sIdx[TOPK];
    __shared__ float sVal[TOPK];
    if (t < TOPK) { sIdx[t] = topIdx[row * TOPK + t]; sVal[t] = topVal[row * TOPK + t]; }
    __syncthreads();

    for (int d = t; d < D_IN; d += 256) {
        const float* wrow = Wd + (size_t)d * D_LATENT;
        float acc = 0.0f;
#pragma unroll
        for (int j = 0; j < TOPK; ++j) acc += sVal[j] * wrow[sIdx[j]];
        Xhat[(size_t)row * D_IN + d] = acc;
    }
}

// ---------------- launch ----------------
extern "C" void kernel_launch(void* const* d_in, const int* in_sizes, int n_in,
                              void* d_out, int out_size, void* d_ws, size_t ws_size,
                              hipStream_t stream) {
    (void)in_sizes; (void)n_in; (void)out_size; (void)ws_size;

    const float* x    = (const float*)d_in[0];   // [BATCH, D_IN]
    const float* Wenc = (const float*)d_in[1];   // [D_LATENT, D_IN]
    const float* Wdec = (const float*)d_in[2];   // [D_IN, D_LATENT]
    // d_in[3] = k (fixed at 32)

    float* xhat = (float*)d_out;                              // [BATCH, D_IN]
    float* z    = (float*)d_out + (size_t)BATCH * D_IN;       // [BATCH, D_LATENT]

    char* ws = (char*)d_ws;
    __bf16* xb = (__bf16*)ws;                                        // 12.6 MB
    __bf16* wb = (__bf16*)(ws + (size_t)BATCH * D_IN * 2);           // 25.2 MB
    size_t off = (size_t)BATCH * D_IN * 2 + (size_t)D_LATENT * D_IN * 2;
    off = (off + 255) & ~(size_t)255;
    int*   topIdx = (int*)(ws + off);                                // 1 MB
    float* topVal = (float*)(ws + off + (size_t)BATCH * TOPK * sizeof(int)); // 1 MB

    cvt_f32_bf16<<<2048, 256, 0, stream>>>(x,    xb, BATCH * D_IN);
    cvt_f32_bf16<<<4096, 256, 0, stream>>>(Wenc, wb, D_LATENT * D_IN);

    dim3 grid(D_LATENT / BN, BATCH / BM);   // 128 x 64 blocks
    sae_encoder_wmma<<<grid, 256, 0, stream>>>(xb, wb, z);

    sae_topk<<<BATCH, 256, 0, stream>>>(z, topIdx, topVal);
    sae_decoder<<<BATCH, 256, 0, stream>>>(Wdec, topIdx, topVal, xhat);
}
